// DEC_26139170963600
// MI455X (gfx1250) — matrix operand order repaired and checked by reading, not compiled, not run on hardware
//
#include <hip/hip_runtime.h>
#include <hip/hip_bf16.h>

typedef __attribute__((ext_vector_type(16))) __bf16 v16bf;
typedef __attribute__((ext_vector_type(8)))  float  v8f;

static __device__ __forceinline__ unsigned bf16bits(float f) {
    unsigned u = __builtin_bit_cast(unsigned, f);
    // round-to-nearest-even bf16, return bits in [15:0]
    unsigned r = u + 0x7FFFu + ((u >> 16) & 1u);
    return r >> 16;
}

static __device__ __forceinline__ __bf16 f2bf(float f) {
    unsigned short h = (unsigned short)bf16bits(f);
    return __builtin_bit_cast(__bf16, h);
}

static __device__ __forceinline__ float bf2f(__bf16 b) {
    unsigned u = (unsigned)__builtin_bit_cast(unsigned short, b) << 16;
    return __builtin_bit_cast(float, u);
}

// ---------------------------------------------------------------------------
// Weight pre-pack: fp32 (O,C,KW) -> bf16 fragments laid out exactly as the
// 16-bit A-matrix VGPR layout (CDNA5 ISA 7.12.2), so each wave's A
// fragment per WMMA is a contiguous 32B vector load per lane.
// Layout: [otile(8)][kkstep(KDIM/32)][lane(32)][e(16)]
// ---------------------------------------------------------------------------
template <int KW>
__global__ __launch_bounds__(256) void pack_w(const float* __restrict__ w,
                                              __bf16* __restrict__ wp) {
    constexpr int C    = 128;
    constexpr int KDIM = C * KW;
    constexpr int NKK  = KDIM / 32;
    const int total = 8 * NKK * 32 * 16;   // == 128 * KDIM
    int idx = blockIdx.x * 256 + threadIdx.x;
    if (idx >= total) return;
    int e      = idx & 15;
    int lane   = (idx >> 4) & 31;
    int kkstep = (idx >> 9) % NKK;
    int otile  = idx / (NKK * 512);
    int m    = lane & 15;
    int half = lane >> 4;
    // element e -> K offset within the 32-wide K step (A-matrix 16-bit layout)
    int ke   = (e < 8 ? e : e + 8) + half * 8;
    int kidx = kkstep * 32 + ke;
    int c = kidx / KW;
    int k = kidx % KW;
    int o = otile * 16 + m;
    wp[idx] = f2bf(w[((size_t)o * C + c) * KW + k]);
}

// ---------------------------------------------------------------------------
// Implicit-GEMM conv1d (VALID, stride S) via v_wmma_f32_16x16x32_bf16.
// Block: 256 threads (8 waves). Each block: one sample n, 16 output positions.
//
// Staging does the im2col directly into the dense 32x16 B-matrix VGPR layout
// (lane<16: K 0-7,16-23, lane>=16: K 8-15,24-31; column = lane&15), packing 8 bf16 into
// registers per item and emitting one ds_store_b128. TIN=float rounds on the
// fly (layer 1); TIN=__bf16 is a raw u16 copy (layers 2-4). The fragment
// buffer is shared by all 8 waves, so the inner loop per WMMA is just one
// contiguous 32B global load (A) + one contiguous 32B LDS load (B) + WMMA.
// Outputs are stored bf16 (consumed by the next bf16 WMMA anyway).
// ---------------------------------------------------------------------------
template <typename TIN, int KW, int STRIDE, int LIN, int LOUT, bool RELU>
__global__ __launch_bounds__(256) void conv_wmma(const TIN* __restrict__ in,
                                                 const __bf16* __restrict__ wpack,
                                                 const float* __restrict__ bias,
                                                 __bf16* __restrict__ out) {
    constexpr int C    = 128;
    constexpr int KDIM = C * KW;
    constexpr int NKK  = KDIM / 32;
    constexpr int FRAG = NKK * 512;        // KDIM * 16 bf16 elements
    static_assert(KDIM % 32 == 0, "K depth must divide 32");

    __shared__ __align__(32) __bf16 bs[FRAG];

    const int n  = blockIdx.y;
    const int l0 = blockIdx.x * 16;
    const TIN* xin = in + (size_t)n * C * LIN;
    const int base = l0 * STRIDE;

    // ---- im2col stage: global -> bf16 B fragments in LDS ----
    for (int t = threadIdx.x; t < FRAG / 8; t += 256) {
        const int idx8   = t * 8;
        const int e0     = idx8 & 15;          // 0 or 8
        const int lane2  = (idx8 >> 4) & 31;
        const int kkstep = idx8 >> 9;
        const int j2     = lane2 & 15;
        const int half2  = lane2 >> 4;
        const int kbase  = kkstep * 32 + half2 * 8 + e0 * 2;
        const int col    = base + j2 * STRIDE;
        unsigned pk[4];
        #pragma unroll
        for (int p = 0; p < 4; ++p) {
            unsigned bits[2];
            #pragma unroll
            for (int s = 0; s < 2; ++s) {
                int kidx = kbase + 2 * p + s;
                int c = kidx / KW, k = kidx % KW;
                int li = col + k;
                if constexpr (__is_same(TIN, float)) {
                    float v = (li < LIN) ? xin[(size_t)c * LIN + li] : 0.0f;
                    bits[s] = bf16bits(v);
                } else {
                    bits[s] = (li < LIN)
                        ? (unsigned)__builtin_bit_cast(unsigned short,
                                                       xin[(size_t)c * LIN + li])
                        : 0u;
                }
            }
            pk[p] = bits[0] | (bits[1] << 16);
        }
        uint4* dst = reinterpret_cast<uint4*>(&bs[idx8]);
        *dst = make_uint4(pk[0], pk[1], pk[2], pk[3]);
    }
    __syncthreads();

    const int wave = threadIdx.x >> 5;   // 0..7 -> output-channel tile
    const int lane = threadIdx.x & 31;
    const int half = lane >> 4;
    const int j    = lane & 15;          // output position within tile (N col)

    v8f acc = {};
    const __bf16* wp = wpack + ((size_t)wave * NKK * 32 + lane) * 16;
    const __bf16* bp = bs + (size_t)lane * 16;

    #pragma unroll 2
    for (int kkstep = 0; kkstep < NKK; ++kkstep) {
        v16bf a = *(const v16bf*)(wp + (size_t)kkstep * 512);
        v16bf b = *(const v16bf*)(bp + (size_t)kkstep * 512);
        acc = __builtin_amdgcn_wmma_f32_16x16x32_bf16(
            false, a, false, b, (short)0, acc, false, false);
    }

    // C/D layout: VGPR r, lane -> M = r + 8*half, N = lane&15
    const int o0 = wave * 16;
    __bf16* op = out + (size_t)n * C * LOUT;
    const int l = l0 + j;
    if (l < LOUT) {
        #pragma unroll
        for (int r = 0; r < 8; ++r) {
            int o = o0 + r + half * 8;
            float v = acc[r] + bias[o];
            if (RELU) v = (v > 0.0f) ? v : 0.1f * v;
            op[(size_t)o * LOUT + l] = f2bf(v);
        }
    }
}

// ---------------------------------------------------------------------------
// Soft-assignment: d2 = ||z - c||^2 (direct, accumulated fp32; z is bf16,
// widened exactly), q = 1/(1+d2) (alpha=1 => exponent is 1), row-normalized.
// One block per sample n.
// ---------------------------------------------------------------------------
__global__ __launch_bounds__(256) void dec_q(const __bf16* __restrict__ z,
                                             const float* __restrict__ centers,
                                             float* __restrict__ q) {
    constexpr int D  = 128 * 59;   // 7552
    constexpr int KC = 64;
    __shared__ float red[256];
    __shared__ float qv[KC + 1];

    const int n = blockIdx.x;
    const __bf16* zp = z + (size_t)n * D;

    for (int kc = 0; kc < KC; ++kc) {
        const float* cp = centers + (size_t)kc * D;
        float s = 0.0f;
        for (int t = threadIdx.x; t < D; t += 256) {
            float d = bf2f(zp[t]) - cp[t];
            s = fmaf(d, d, s);
        }
        red[threadIdx.x] = s;
        __syncthreads();
        for (int off = 128; off > 0; off >>= 1) {
            if (threadIdx.x < off) red[threadIdx.x] += red[threadIdx.x + off];
            __syncthreads();
        }
        if (threadIdx.x == 0) qv[kc] = 1.0f / (1.0f + red[0]);
        __syncthreads();
    }
    if (threadIdx.x == 0) {
        float sum = 0.0f;
        for (int i = 0; i < KC; ++i) sum += qv[i];
        qv[KC] = sum;
    }
    __syncthreads();
    if (threadIdx.x < KC)
        q[(size_t)n * KC + threadIdx.x] = qv[threadIdx.x] / qv[KC];
}

// ---------------------------------------------------------------------------
// Launch: pack weights -> conv x4 (ping-pong bf16 activations in ws) -> q
// ws layout (bytes), all activations bf16:
//   A = 0          : h1 (33,095,680) later reused for h3 (7,929,856)
//   B = 33,095,680 : h2 (16,187,392) later reused for z/h4 (3,866,624)
//   W = 49,283,072 : packed bf16 weights (1,245,184)
// ---------------------------------------------------------------------------
extern "C" void kernel_launch(void* const* d_in, const int* in_sizes, int n_in,
                              void* d_out, int out_size, void* d_ws, size_t ws_size,
                              hipStream_t stream) {
    const float* x  = (const float*)d_in[0];
    const float* w1 = (const float*)d_in[1]; const float* b1 = (const float*)d_in[2];
    const float* w2 = (const float*)d_in[3]; const float* b2 = (const float*)d_in[4];
    const float* w3 = (const float*)d_in[5]; const float* b3 = (const float*)d_in[6];
    const float* w4 = (const float*)d_in[7]; const float* b4 = (const float*)d_in[8];
    const float* centers = (const float*)d_in[9];

    const size_t OFF_A = 0;
    const size_t OFF_B = (size_t)256 * 128 * 505 * 2;           // 33,095,680
    const size_t OFF_W = OFF_B + (size_t)256 * 128 * 247 * 2;   // 49,283,072

    __bf16* bufA = (__bf16*)((char*)d_ws + OFF_A);
    __bf16* bufB = (__bf16*)((char*)d_ws + OFF_B);
    __bf16* wp1  = (__bf16*)((char*)d_ws + OFF_W);
    __bf16* wp2  = wp1 + (size_t)128 * 128 * 15;
    __bf16* wp3  = wp2 + (size_t)128 * 128 * 12;
    __bf16* wp4  = wp3 + (size_t)128 * 128 * 7;

    // weight pre-pack (bf16, WMMA A-fragment order)
    pack_w<15><<<(128 * 128 * 15 + 255) / 256, 256, 0, stream>>>(w1, wp1);
    pack_w<12><<<(128 * 128 * 12 + 255) / 256, 256, 0, stream>>>(w2, wp2);
    pack_w< 7><<<(128 * 128 *  7 + 255) / 256, 256, 0, stream>>>(w3, wp3);
    pack_w< 4><<<(128 * 128 *  4 + 255) / 256, 256, 0, stream>>>(w4, wp4);

    // encoder: 1024 -> 505 -> 247 -> 121 -> 59
    conv_wmma<float,  15, 2, 1024, 505, true ><<<dim3(32, 256), 256, 0, stream>>>(x,    wp1, b1, bufA);
    conv_wmma<__bf16, 12, 2,  505, 247, true ><<<dim3(16, 256), 256, 0, stream>>>(bufA, wp2, b2, bufB);
    conv_wmma<__bf16,  7, 2,  247, 121, true ><<<dim3( 8, 256), 256, 0, stream>>>(bufB, wp3, b3, bufA);
    conv_wmma<__bf16,  4, 2,  121,  59, false><<<dim3( 4, 256), 256, 0, stream>>>(bufA, wp4, b4, bufB);

    // soft assignments
    dec_q<<<256, 256, 0, stream>>>(bufB, centers, (float*)d_out);
}
